// TreeGenerator_16698832847125
// MI455X (gfx1250) — compile-verified, matrix operand order checked by reading, not run on hardware
//
#include <hip/hip_runtime.h>

typedef __attribute__((ext_vector_type(16))) __bf16 v16bf;
typedef __attribute__((ext_vector_type(8)))  float  v8f;

#define S_LEN 16384
#define HID   512
#define EMB   512
#define RULES 2048
#define G3    1536   // 3*HID

union FragU { v16bf v; uint4 u[2]; };

__device__ __forceinline__ void load_frag(FragU& f, const __bf16* p) {
  f.u[0] = *(const uint4*)(p);
  f.u[1] = *(const uint4*)(p + 16);
}

// ---------------- f32 -> bf16 conversion ----------------
__global__ void cvt_bf16_kernel(const float* __restrict__ src,
                                __bf16* __restrict__ dst, int n) {
  int i = blockIdx.x * blockDim.x + threadIdx.x;
  if (i < n) dst[i] = (__bf16)src[i];
}

// ---------------- embedding gather (teacher-forced shift) ----------------
__global__ void gather_kernel(const float* __restrict__ emb,
                              const int* __restrict__ acts,
                              const int* __restrict__ pacts,
                              __bf16* __restrict__ Xb) {
  const int s = blockIdx.x;
  const int a = (s == 0) ? 0 : (acts[s - 1] + 1);  // PAD(-1)+OFF(1)=0 row (zeroed)
  const int p = pacts[s] + 1;
  const int c = threadIdx.x * 2;
  const float2 av = *(const float2*)(emb + (size_t)a * EMB + c);
  const float2 pv = *(const float2*)(emb + (size_t)p * EMB + c);
  __bf16* row = Xb + (size_t)s * (2 * EMB);
  row[c]           = (__bf16)av.x;  row[c + 1]       = (__bf16)av.y;
  row[EMB + c]     = (__bf16)pv.x;  row[EMB + c + 1] = (__bf16)pv.y;
}

// ---------------- WMMA GEMM: C[M,N] = A[M,K] * W[N,K]^T + bias[N] ----------
// Block = 256 thr (8 waves), WG tile 128x128. Each wave: 4 M-tiles x 2 N-tiles.
// K unrolled by 64 with two fragment buffers so loads for the next K-step are
// in flight while the current 8 WMMAs issue (avoids s_wait_loadcnt 0 stalls).
__global__ __launch_bounds__(256)
void gemm_bf16_kernel(const __bf16* __restrict__ A, const __bf16* __restrict__ W,
                      const float* __restrict__ bias, float* __restrict__ C,
                      int N, int K) {
  const int lane  = threadIdx.x & 31;
  const int wave  = threadIdx.x >> 5;
  const int mr    = lane & 15;     // row/col within tile
  const int hx    = lane >> 4;     // half-wave selector
  const int mBase = blockIdx.x * 128 + (wave & 1) * 64;
  const int nBase = blockIdx.y * 128 + (wave >> 1) * 32;

  const __bf16* aRow[4];
  const __bf16* bRow[2];
#pragma unroll
  for (int t = 0; t < 4; ++t)
    aRow[t] = A + (size_t)(mBase + t * 16 + mr) * K + hx * 8;
#pragma unroll
  for (int u = 0; u < 2; ++u)
    bRow[u] = W + (size_t)(nBase + u * 16 + mr) * K + hx * 8;

  v8f acc[4][2];
#pragma unroll
  for (int t = 0; t < 4; ++t)
#pragma unroll
    for (int u = 0; u < 2; ++u)
      acc[t][u] = (v8f){0.f,0.f,0.f,0.f,0.f,0.f,0.f,0.f};

  FragU a0[4], b0[2], a1[4], b1[2];
#pragma unroll
  for (int t = 0; t < 4; ++t) load_frag(a0[t], aRow[t]);
#pragma unroll
  for (int u = 0; u < 2; ++u) load_frag(b0[u], bRow[u]);

  for (int k = 0; k < K; k += 64) {
    // prefetch K-step k+32 while computing on k
#pragma unroll
    for (int t = 0; t < 4; ++t) load_frag(a1[t], aRow[t] + k + 32);
#pragma unroll
    for (int u = 0; u < 2; ++u) load_frag(b1[u], bRow[u] + k + 32);
#pragma unroll
    for (int t = 0; t < 4; ++t)
#pragma unroll
      for (int u = 0; u < 2; ++u)
        acc[t][u] = __builtin_amdgcn_wmma_f32_16x16x32_bf16(
            false, a0[t].v, false, b0[u].v, (short)0, acc[t][u], false, false);
    if (k + 64 < K) {
#pragma unroll
      for (int t = 0; t < 4; ++t) load_frag(a0[t], aRow[t] + k + 64);
#pragma unroll
      for (int u = 0; u < 2; ++u) load_frag(b0[u], bRow[u] + k + 64);
    }
#pragma unroll
    for (int t = 0; t < 4; ++t)
#pragma unroll
      for (int u = 0; u < 2; ++u)
        acc[t][u] = __builtin_amdgcn_wmma_f32_16x16x32_bf16(
            false, a1[t].v, false, b1[u].v, (short)0, acc[t][u], false, false);
  }

#pragma unroll
  for (int u = 0; u < 2; ++u) {
    const int col = nBase + u * 16 + mr;
    const float bv = bias[col];
#pragma unroll
    for (int t = 0; t < 4; ++t) {
#pragma unroll
      for (int r = 0; r < 8; ++r) {
        const int row = mBase + t * 16 + hx * 8 + r;  // C VGPR r: M=r / M=r+8
        C[(size_t)row * N + col] = acc[t][u][r] + bv;
      }
    }
  }
}

// ---------------- scan init: seed h buffers, zero barriers --------------
__global__ void scan_init_kernel(const float* __restrict__ init_state,
                                 float* hbufA, float* hbufB, unsigned* ctrs) {
  const int i = threadIdx.x;
  if (i < HID) {
    hbufA[i] = init_state[i];         // layer 0 h0
    hbufB[i] = init_state[HID + i];   // layer 1 h0
  }
  if (i == 0) { ctrs[0] = 0u; ctrs[64] = 0u; }
}

// ---------------- GRU recurrent scan --------------------------------------
// 16 workgroups; WG b owns hidden indices [b*32, b*32+32). Its 3x32 rows of
// W_hh (192 KB f32) stay resident in LDS for the entire sequence. Per step:
// compute 32 (r,z,n) dot products, publish 32 new h values via L2, then a
// grid-wide monotonic atomic barrier (s_sleep backoff) releases step t+1.
__global__ __launch_bounds__(256)
void gru_scan_kernel(const float* __restrict__ W_hh, const float* __restrict__ b_hh,
                     const float* __restrict__ XI, __bf16* __restrict__ outB,
                     float* hbuf, unsigned* ctr) {
  extern __shared__ float wl[];            // [3][32][512] f32 = 192 KB
  const int tid  = threadIdx.x;
  const int base = blockIdx.x * 32;

  for (int e = tid; e < 3 * 32 * 512; e += 256) {
    const int g   = e / (32 * 512);
    const int rem = e % (32 * 512);
    const int idx = rem / 512;
    const int k   = rem % 512;
    wl[e] = W_hh[(size_t)(g * HID + base + idx) * HID + k];
  }
  __syncthreads();

  const int lane = tid & 31;
  const int wave = tid >> 5;     // 8 waves x 4 indices = 32 indices per WG

  float bh[4][3];
#pragma unroll
  for (int q = 0; q < 4; ++q) {
    const int i = base + wave * 4 + q;
#pragma unroll
    for (int g = 0; g < 3; ++g) bh[q][g] = b_hh[g * HID + i];
  }

  const unsigned nwg = gridDim.x;
  for (int t = 0; t < S_LEN; ++t) {
    // pull next step's XI slice toward the WGP while we work on this step
    if (t + 1 < S_LEN && tid < 3)
      __builtin_prefetch(XI + (size_t)(t + 1) * G3 + tid * HID + base, 0, 0);

    volatile float* hcur = hbuf + ((t) & 1) * HID;
    volatile float* hnxt = hbuf + ((t + 1) & 1) * HID;
    float hreg[16];
#pragma unroll
    for (int j = 0; j < 16; ++j) hreg[j] = hcur[lane * 16 + j];

#pragma unroll
    for (int q = 0; q < 4; ++q) {
      const int i = base + wave * 4 + q;
      const float* wr = wl + ((0 * 32 + wave * 4 + q) * 512) + lane * 16;
      const float* wz = wl + ((1 * 32 + wave * 4 + q) * 512) + lane * 16;
      const float* wn = wl + ((2 * 32 + wave * 4 + q) * 512) + lane * 16;
      float sr = 0.f, sz = 0.f, sn = 0.f;
#pragma unroll
      for (int j = 0; j < 16; ++j) {
        sr = fmaf(wr[j], hreg[j], sr);
        sz = fmaf(wz[j], hreg[j], sz);
        sn = fmaf(wn[j], hreg[j], sn);
      }
#pragma unroll
      for (int off = 16; off > 0; off >>= 1) {
        sr += __shfl_xor(sr, off);
        sz += __shfl_xor(sz, off);
        sn += __shfl_xor(sn, off);
      }
      if (lane == q) {
        const float* xirow = XI + (size_t)t * G3;
        const float xr = xirow[i];
        const float xz = xirow[HID + i];
        const float xn = xirow[2 * HID + i];
        const float hr = sr + bh[q][0];
        const float hz = sz + bh[q][1];
        const float hn = sn + bh[q][2];
        const float rg = 1.f / (1.f + __expf(-(xr + hr)));
        const float zg = 1.f / (1.f + __expf(-(xz + hz)));
        const float ng = tanhf(xn + rg * hn);     // torch semantics: r*(Wh+b)
        const float ho = hcur[i];
        const float hv = (1.f - zg) * ng + zg * ho;
        hnxt[i] = hv;
        outB[(size_t)t * HID + i] = (__bf16)hv;
      }
    }
    __threadfence();
    __syncthreads();
    if (tid == 0) {
      atomicAdd(ctr, 1u);
      const unsigned target = (unsigned)(t + 1) * nwg;
      while (__hip_atomic_load(ctr, __ATOMIC_ACQUIRE, __HIP_MEMORY_SCOPE_AGENT) < target) {
        __builtin_amdgcn_s_sleep(1);
      }
    }
    __syncthreads();
  }
}

// ---------------- in-place row log-softmax over RULES columns --------------
__global__ __launch_bounds__(256)
void logsoftmax_kernel(float* __restrict__ logits) {
  __shared__ float red[8];
  const int s = blockIdx.x;
  float* row = logits + (size_t)s * RULES;
  const int tid = threadIdx.x;

  float m = -3.4e38f;
  for (int i = tid; i < RULES; i += 256) m = fmaxf(m, row[i]);
#pragma unroll
  for (int off = 16; off > 0; off >>= 1) m = fmaxf(m, __shfl_xor(m, off));
  if ((tid & 31) == 0) red[tid >> 5] = m;
  __syncthreads();
  float mm = red[0];
#pragma unroll
  for (int w = 1; w < 8; ++w) mm = fmaxf(mm, red[w]);
  __syncthreads();

  float sum = 0.f;
  for (int i = tid; i < RULES; i += 256) sum += __expf(row[i] - mm);
#pragma unroll
  for (int off = 16; off > 0; off >>= 1) sum += __shfl_xor(sum, off);
  if ((tid & 31) == 0) red[tid >> 5] = sum;
  __syncthreads();
  float ss = 0.f;
#pragma unroll
  for (int w = 0; w < 8; ++w) ss += red[w];

  const float lse = mm + __logf(ss);
  for (int i = tid; i < RULES; i += 256) row[i] = row[i] - lse;
}

// ---------------- value head: v[s] = out[s] . value_W + value_b -----------
__global__ void value_kernel(const __bf16* __restrict__ outB,
                             const float* __restrict__ vW,
                             const float* __restrict__ vb,
                             float* __restrict__ vals) {
  const int s = blockIdx.x * blockDim.x + threadIdx.x;
  if (s >= S_LEN) return;
  const __bf16* h = outB + (size_t)s * HID;
  float acc = 0.f;
  for (int k = 0; k < HID; ++k) acc = fmaf((float)h[k], vW[k], acc);
  vals[s] = acc + vb[0];
}

// ---------------------------------------------------------------------------
extern "C" void kernel_launch(void* const* d_in, const int* in_sizes, int n_in,
                              void* d_out, int out_size, void* d_ws, size_t ws_size,
                              hipStream_t stream) {
  (void)in_sizes; (void)n_in; (void)out_size; (void)ws_size;
  const float* init_state = (const float*)d_in[0];
  const float* emb   = (const float*)d_in[1];
  const float* Wih0  = (const float*)d_in[2];
  const float* Whh0  = (const float*)d_in[3];
  const float* bih0  = (const float*)d_in[4];
  const float* bhh0  = (const float*)d_in[5];
  const float* Wih1  = (const float*)d_in[6];
  const float* Whh1  = (const float*)d_in[7];
  const float* bih1  = (const float*)d_in[8];
  const float* bhh1  = (const float*)d_in[9];
  const float* actW  = (const float*)d_in[10];
  const float* actb  = (const float*)d_in[11];
  const float* valW  = (const float*)d_in[12];
  const float* valb  = (const float*)d_in[13];
  const int*   acts  = (const int*)d_in[14];
  const int*   pacts = (const int*)d_in[15];
  float* out = (float*)d_out;

  char* ws = (char*)d_ws;
  size_t off = 0;
  auto alloc = [&](size_t bytes) -> void* {
    void* p = ws + off;
    off = (off + bytes + 255) & ~(size_t)255;
    return p;
  };
  __bf16* Xb    = (__bf16*)alloc((size_t)S_LEN * 1024 * 2);  // gathered input
  float*  XI    = (float*) alloc((size_t)S_LEN * G3 * 4);    // x@W_ih^T+b (reused L0/L1)
  __bf16* H0b   = (__bf16*)alloc((size_t)S_LEN * HID * 2);   // layer-0 output
  __bf16* OUTb  = (__bf16*)alloc((size_t)S_LEN * HID * 2);   // layer-1 output
  __bf16* Wih0b = (__bf16*)alloc((size_t)G3 * 1024 * 2);
  __bf16* Wih1b = (__bf16*)alloc((size_t)G3 * HID * 2);
  __bf16* actWb = (__bf16*)alloc((size_t)RULES * HID * 2);
  float*  hbufA = (float*) alloc(2 * HID * 4);
  float*  hbufB = (float*) alloc(2 * HID * 4);
  unsigned* ctrs = (unsigned*)alloc(256);                    // [0]=L0, [64]=L1

  const size_t LDS_SCAN = (size_t)3 * 32 * 512 * 4;          // 192 KB

  // 1) weight conversions to bf16
  cvt_bf16_kernel<<<(G3 * 1024 + 255) / 256, 256, 0, stream>>>(Wih0, Wih0b, G3 * 1024);
  cvt_bf16_kernel<<<(G3 * HID  + 255) / 256, 256, 0, stream>>>(Wih1, Wih1b, G3 * HID);
  cvt_bf16_kernel<<<(RULES * HID + 255) / 256, 256, 0, stream>>>(actW, actWb, RULES * HID);

  // 2) embedding gather with teacher-forced shift
  gather_kernel<<<S_LEN, 256, 0, stream>>>(emb, acts, pacts, Xb);

  // 3) XI0 = X @ W_ih0^T + b_ih0
  gemm_bf16_kernel<<<dim3(S_LEN / 128, G3 / 128), 256, 0, stream>>>(
      Xb, Wih0b, bih0, XI, G3, 1024);

  // 4) seed h buffers + barrier counters
  scan_init_kernel<<<1, 512, 0, stream>>>(init_state, hbufA, hbufB, ctrs);

  // 5) layer-0 recurrent scan
  gru_scan_kernel<<<16, 256, LDS_SCAN, stream>>>(Whh0, bhh0, XI, H0b, hbufA, ctrs + 0);

  // 6) XI1 = H0 @ W_ih1^T + b_ih1 (reuse XI buffer)
  gemm_bf16_kernel<<<dim3(S_LEN / 128, G3 / 128), 256, 0, stream>>>(
      H0b, Wih1b, bih1, XI, G3, HID);

  // 7) layer-1 recurrent scan
  gru_scan_kernel<<<16, 256, LDS_SCAN, stream>>>(Whh1, bhh1, XI, OUTb, hbufB, ctrs + 64);

  // 8) logits = OUT @ action_W^T + action_b, directly into d_out
  gemm_bf16_kernel<<<dim3(S_LEN / 128, RULES / 128), 256, 0, stream>>>(
      OUTb, actWb, actb, out, RULES, HID);

  // 9) in-place log-softmax per row
  logsoftmax_kernel<<<S_LEN, 256, 0, stream>>>(out);

  // 10) value head
  value_kernel<<<(S_LEN + 255) / 256, 256, 0, stream>>>(
      OUTb, valW, valb, out + (size_t)S_LEN * RULES);
}
